// GrokkingSNN_50165218017467
// MI455X (gfx1250) — matrix operand order repaired and checked by reading, not compiled, not run on hardware
//
#include <hip/hip_runtime.h>

// ---------------------------------------------------------------------------
// Fused SNN forward for MI455X (gfx1250, wave32, bf16 WMMA 16x16x32)
// ---------------------------------------------------------------------------

typedef __attribute__((ext_vector_type(16))) __bf16 v16bf;
typedef __attribute__((ext_vector_type(8)))  __bf16 v8bf;
typedef __attribute__((ext_vector_type(8)))  float  v8f;
typedef __attribute__((ext_vector_type(4)))  unsigned int u32x4;
typedef __attribute__((ext_vector_type(8)))  int i32x8;
typedef __attribute__((ext_vector_type(4)))  int i32x4;

#define HID      512
#define TWO_HID  1024
#define PDIM     97
#define NPAD     112
#define NSTEPS   15
#define BROWS    16
#define W2STRIDE 520   // 512 halfs + 8 halfs (16B) pad per row -> matches TDM pad
#define SPKSTRIDE 520

static __device__ __forceinline__ __bf16 f2bf(float f) {
  union { float f; unsigned u; } v; v.f = f;
  unsigned r = v.u + 0x7FFFu + ((v.u >> 16) & 1u);   // round-to-nearest-even
  union { unsigned short s; __bf16 b; } o; o.s = (unsigned short)(r >> 16);
  return o.b;
}

static __device__ __forceinline__ v16bf cat16(v8bf lo, v8bf hi) {
  return __builtin_shufflevector(lo, hi, 0,1,2,3,4,5,6,7,8,9,10,11,12,13,14,15);
}

// ---------------- prepass: f32 -> bf16 conversion into d_ws ----------------
// W1bf [512][1024], W2bf [112][512] (zero-padded rows 97..111), Ebf [97][512]
__global__ void cvt_weights(const float* __restrict__ W1,
                            const float* __restrict__ W2,
                            const float* __restrict__ E,
                            __bf16* __restrict__ W1bf,
                            __bf16* __restrict__ W2bf,
                            __bf16* __restrict__ Ebf) {
  int i = blockIdx.x * 256 + threadIdx.x;
  if (i < HID * TWO_HID) W1bf[i] = f2bf(W1[i]);
  if (i < NPAD * HID)    W2bf[i] = (i < PDIM * HID) ? f2bf(W2[i]) : f2bf(0.0f);
  if (i < PDIM * HID)    Ebf[i]  = f2bf(E[i]);
}

// ---------------- fused SNN kernel -----------------------------------------
__global__ __launch_bounds__(256)
void snn_fused(const int*   __restrict__ x,
               const float* __restrict__ b1,
               const float* __restrict__ b2,
               const float* __restrict__ beta1p,
               const float* __restrict__ beta2p,
               const float* __restrict__ thr1p,
               const __bf16* __restrict__ W1bf,
               const __bf16* __restrict__ W2bf,
               const __bf16* __restrict__ Ebf,
               float* __restrict__ out) {
  __shared__ __align__(16) __bf16 sW2[NPAD * W2STRIDE];    // 116,480 B
  __shared__ __align__(16) __bf16 sSpk[BROWS * SPKSTRIDE]; // 16,640 B

  const int tid  = threadIdx.x;
  const int wave = tid >> 5;
  const int lane = tid & 31;
  const int lm   = lane & 15;
  const int hi   = lane >> 4;

  // ---- stage W2 (bf16, 112 padded rows) into LDS via Tensor Data Mover ----
#if __has_builtin(__builtin_amdgcn_tensor_load_to_lds)
  if (wave == 0) {
    // D#: 2D tensor 112 x 128 (8-byte units), LDS pad of 4 DWORDs after every
    // 256 DWORDs (one 1024B row) -> LDS row stride 1040B (bank de-conflict).
    unsigned lds_addr = (unsigned)(unsigned long long)(&sW2[0]);
    unsigned long long ga = (unsigned long long)W2bf;
    u32x4 g0;
    g0[0] = 1u;                                   // count=1, user descriptor
    g0[1] = lds_addr;                             // lds_addr
    g0[2] = (unsigned)ga;                         // global_addr[31:0]
    g0[3] = (unsigned)((ga >> 32) & 0x01FFFFFFull) | 0x80000000u; // addr[56:32], type=2
    i32x8 g1;
    g1[0] = (int)((3u << 16) | (1u << 20) | (7u << 22) | (3u << 25));
             // data_size=8B, pad_enable, pad_interval=256dw, pad_amount=4dw
    g1[1] = (int)(128u << 16);                    // tensor_dim0 = 128 (8B units)
    g1[2] = (int)(112u << 16);                    // tensor_dim1 = 112
    g1[3] = (int)(128u << 16);                    // tile_dim0   = 128
    g1[4] = 112;                                  // tile_dim1   = 112
    g1[5] = 128;                                  // tensor_dim0_stride = 128
    g1[6] = 0;
    g1[7] = 0;
    i32x4 gz = {0, 0, 0, 0};
#if __clang_major__ >= 23
    i32x8 gz8 = {0, 0, 0, 0, 0, 0, 0, 0};
    __builtin_amdgcn_tensor_load_to_lds(g0, g1, gz, gz, gz8, 0);
#else
    __builtin_amdgcn_tensor_load_to_lds(g0, g1, gz, gz, 0);
#endif
#if __has_builtin(__builtin_amdgcn_s_wait_tensorcnt)
    __builtin_amdgcn_s_wait_tensorcnt(0);
#else
    asm volatile("s_wait_tensorcnt 0x0" ::: "memory");
#endif
  }
#else
  for (int i = tid; i < NPAD * (HID / 8); i += 256) {
    int r = i >> 6, c = (i & 63) << 3;
    *(v8bf*)&sW2[r * W2STRIDE + c] = *(const v8bf*)&W2bf[r * HID + c];
  }
#endif
  __syncthreads();

  // ---- scalar parameter clamps -------------------------------------------
  const float beta1 = fminf(fmaxf(beta1p[0], 0.1f), 0.9f);
  const float beta2 = fminf(fmaxf(beta2p[0], 0.1f), 0.9f);
  const float thr1  = fmaxf(thr1p[0], 0.1f);

  // ---- pre-converted bf16 embedding rows for this lane's batch row --------
  const int brow = blockIdx.x * BROWS + lm;
  const int ia = x[brow * 2 + 0];
  const int ib = x[brow * 2 + 1];
  const __bf16* rowA = Ebf + (long long)ia * HID;
  const __bf16* rowB = Ebf + (long long)ib * HID;

  // ---- GEMM1: cur1 = emb @ W1^T + b1 (wave owns 64 hidden columns) --------
  const v8f vzero = {0.f, 0.f, 0.f, 0.f, 0.f, 0.f, 0.f, 0.f};
  v8f acc1[4] = {vzero, vzero, vzero, vzero};
  const int nbase = wave * 64;

  for (int kb = 0; kb < TWO_HID; kb += 32) {
    const __bf16* arow = (kb < HID) ? rowA : rowB;
    const int koff = kb & (HID - 1);
    // A 16x32 bf16: lane lm row; lo lanes K {0..7,16..23}, hi lanes {8..15,24..31}
    v16bf amat = cat16(*(const v8bf*)(arow + koff + hi * 8),
                       *(const v8bf*)(arow + koff + 16 + hi * 8));
#pragma unroll
    for (int nt = 0; nt < 4; ++nt) {
      // B 32x16 bf16: lane n holds W1 row n; lo lanes K {0..7,16..23}, hi lanes {8..15,24..31}
      const __bf16* wp = W1bf + (size_t)(nbase + nt * 16 + lm) * TWO_HID + kb + hi * 8;
      v16bf bmat = cat16(*(const v8bf*)wp, *(const v8bf*)(wp + 16));
      acc1[nt] = __builtin_amdgcn_wmma_f32_16x16x32_bf16(
          false, amat, false, bmat, (short)0, acc1[nt], false, false);
    }
  }
#pragma unroll
  for (int nt = 0; nt < 4; ++nt) {
    const float bias = b1[nbase + nt * 16 + lm];
#pragma unroll
    for (int r = 0; r < 8; ++r) acc1[nt][r] += bias;
  }

  // ---- 15-step LIF recurrence --------------------------------------------
  v8f mem1[4] = {vzero, vzero, vzero, vzero};
  v8f mem2 = vzero;
  const int n2 = wave * 16 + lm;                       // output column (waves 0..6)
  const float bias2 = (wave < 7 && n2 < PDIM) ? b2[n2] : 0.0f;
  const __bf16 ONE = f2bf(1.0f), ZER = f2bf(0.0f);

  for (int s = 0; s < NSTEPS; ++s) {
    // LIF1: reset from previous mem, spike from new mem; spikes -> LDS (bf16)
#pragma unroll
    for (int nt = 0; nt < 4; ++nt) {
      const int n = nbase + nt * 16 + lm;
#pragma unroll
      for (int r = 0; r < 8; ++r) {
        float prev = mem1[nt][r];
        float mnew = beta1 * prev + acc1[nt][r] - ((prev > thr1) ? thr1 : 0.0f);
        mem1[nt][r] = mnew;
        sSpk[(r + hi * 8) * SPKSTRIDE + n] = (mnew > thr1) ? ONE : ZER;
      }
    }
    __syncthreads();

    // GEMM2: cur2 = spk @ W2^T; waves 0..6 each own one 16x16 output tile
    if (wave < 7) {
      v8f acc2 = vzero;
#pragma unroll 4
      for (int kb = 0; kb < HID; kb += 32) {
        const __bf16* ap = &sSpk[lm * SPKSTRIDE + kb + hi * 8];
        v16bf amat = cat16(*(const v8bf*)ap, *(const v8bf*)(ap + 16));
        const __bf16* bp = &sW2[n2 * W2STRIDE + kb + hi * 8];
        v16bf bmat = cat16(*(const v8bf*)bp, *(const v8bf*)(bp + 16));
        acc2 = __builtin_amdgcn_wmma_f32_16x16x32_bf16(
            false, amat, false, bmat, (short)0, acc2, false, false);
      }
#pragma unroll
      for (int r = 0; r < 8; ++r)
        mem2[r] = beta2 * mem2[r] + acc2[r] + bias2;   // lif2, reset='none'
    }
    __syncthreads();
  }

  // ---- write final mem2 [B, 97] f32 --------------------------------------
  if (wave < 7 && n2 < PDIM) {
#pragma unroll
    for (int r = 0; r < 8; ++r)
      out[(size_t)(blockIdx.x * BROWS + r + hi * 8) * PDIM + n2] = mem2[r];
  }
}

// ---------------------------------------------------------------------------
extern "C" void kernel_launch(void* const* d_in, const int* in_sizes, int n_in,
                              void* d_out, int out_size, void* d_ws, size_t ws_size,
                              hipStream_t stream) {
  (void)in_sizes; (void)n_in; (void)out_size; (void)ws_size;
  const int*   x       = (const int*)  d_in[0];
  const float* embed_w = (const float*)d_in[1];
  const float* W1      = (const float*)d_in[2];
  const float* b1      = (const float*)d_in[3];
  const float* W2      = (const float*)d_in[4];
  const float* b2      = (const float*)d_in[5];
  const float* beta1   = (const float*)d_in[6];
  const float* beta2   = (const float*)d_in[7];
  const float* thr1    = (const float*)d_in[8];
  // d_in[9] = thr2: unused by the forward output (lif2 has reset='none')

  const size_t w1_bytes = (size_t)HID * TWO_HID * 2;   // 1,048,576
  const size_t w2_bytes = (size_t)NPAD * HID * 2;      //   114,688
  __bf16* W1bf = (__bf16*)d_ws;
  __bf16* W2bf = (__bf16*)((char*)d_ws + w1_bytes);
  __bf16* Ebf  = (__bf16*)((char*)d_ws + w1_bytes + w2_bytes); // 99,328 B

  cvt_weights<<<(HID * TWO_HID + 255) / 256, 256, 0, stream>>>(
      W1, W2, embed_w, W1bf, W2bf, Ebf);

  const int nblocks = 32768 / BROWS;  // 2048 workgroups of 8 wave32
  snn_fused<<<nblocks, 256, 0, stream>>>(x, b1, b2, beta1, beta2, thr1,
                                         W1bf, W2bf, Ebf, (float*)d_out);
}
